// PaiAutoencoder_79491254715013
// MI455X (gfx1250) — compile-verified, matrix operand order checked by reading, not run on hardware
//
#include <hip/hip_runtime.h>
#include <hip/hip_bf16.h>

// ---------------------------------------------------------------------------
// CDNA5 (gfx1250) implementation of the PAI point-cloud autoencoder.
// Heavy GEMMs (pool/unpool with D/U matrices, PAI conv GEMMs, residual GEMMs,
// fc_enc) run on v_wmma_f32_16x16x32_f16 (f16 operands staged in LDS,
// f32 accumulate). HBM floor: streaming ~550MB of fp32 D/U matrices once
// (~24us @ 23.3 TB/s); activations are L2-resident (192MB L2).
// Split-K (deterministic partials + fixed-order reduction) provides the wave
// parallelism needed to saturate HBM: output tiles are few (M/16) but K is
// huge, so grid.y slices the reduction (K-slices rounded to 32).
// ---------------------------------------------------------------------------

typedef __attribute__((ext_vector_type(16))) _Float16 v16h;
typedef __attribute__((ext_vector_type(8)))  _Float16 v8h;
typedef __attribute__((ext_vector_type(8)))  float    v8f;

// Build one lane's A/B fragment for V_WMMA_F32_16X16X32_F16.
// Per the CDNA5 ISA layout, lane (group = lane>>4) holds:
//   elements 0..7  = K  (8*group)   .. (8*group+7)     -> contiguous 16B in LDS
//   elements 8..15 = K  (16+8*group)..(16+8*group+7)   -> contiguous 16B in LDS
__device__ __forceinline__ v16h ld_frag(const _Float16* row32, int group) {
  v8h lo = *(const v8h*)(row32 + 8 * group);
  v8h hi = *(const v8h*)(row32 + 16 + 8 * group);
  v16h v;
#pragma unroll
  for (int e = 0; e < 8; ++e) { v[e] = lo[e]; v[8 + e] = hi[e]; }
  return v;
}

__device__ __forceinline__ float elu_f(float v) {
  return v > 0.f ? v : (__expf(v) - 1.f);
}

// ---------------------------------------------------------------------------
// Split-K GEMM:  Ypart[slice][b] (M x N) = L (M x Kslice) @ X[b] (Kslice x N)
// One wave per (16 output rows, K-slice). BB = batch count, NT = N/16 tiles.
// grid.x = ceil(M/16), grid.y = #K-slices, ks = K per slice (multiple of 32).
// ---------------------------------------------------------------------------
template <int BB, int NT>
__global__ void __launch_bounds__(32)
pool_gemm(const float* __restrict__ L, const float* __restrict__ X,
          float* __restrict__ Y, int M, int Kd, int ks) {
  constexpr int N = NT * 16;
  __shared__ _Float16 As[16 * 32];          // L tile: 16 rows x 32 K
  __shared__ _Float16 Bst[BB * N * 32];     // X tiles, transposed: [(b,n)][32 K]
  const int lane  = threadIdx.x;
  const int group = lane >> 4;
  const int mrow  = lane & 15;
  const int m0    = blockIdx.x * 16;
  const int k0    = blockIdx.y * ks;
  const int k1    = (k0 + ks < Kd) ? (k0 + ks) : Kd;
  float* Yp = Y + (long)blockIdx.y * BB * M * N;   // this slice's partial

  v8f zero = {};
  v8f acc[BB][NT];
#pragma unroll
  for (int b = 0; b < BB; ++b)
#pragma unroll
    for (int t = 0; t < NT; ++t) acc[b][t] = zero;

  for (int kk = k0; kk < k1; kk += 32) {
    // A tile: row i, col lane (coalesced 128B per row); prefetch next K-slab.
#pragma unroll
    for (int i = 0; i < 16; ++i) {
      int gm = m0 + i, gk = kk + lane;
      float v = (gm < M && gk < Kd) ? L[(long)gm * Kd + gk] : 0.f;
      if (gm < M && gk + 32 < Kd)
        __builtin_prefetch(&L[(long)gm * Kd + gk + 32], 0, 0);
      As[i * 32 + lane] = (_Float16)v;
    }
    // B tiles for all batches / N-tiles (coalesced over columns)
    const int TE = 32 * N * BB;
    for (int idx = lane; idx < TE; idx += 32) {
      int b   = idx / (32 * N);
      int rem = idx - b * 32 * N;
      int r   = rem / N;
      int c   = rem - r * N;
      int gk  = kk + r;
      float v = (gk < Kd) ? X[(long)b * Kd * N + (long)gk * N + c] : 0.f;
      Bst[(b * N + c) * 32 + r] = (_Float16)v;
    }
    __syncthreads();
    v16h a = ld_frag(&As[mrow * 32], group);
#pragma unroll
    for (int b = 0; b < BB; ++b)
#pragma unroll
      for (int t = 0; t < NT; ++t) {
        v16h bf = ld_frag(&Bst[(b * N + t * 16 + mrow) * 32], group);
        acc[b][t] = __builtin_amdgcn_wmma_f32_16x16x32_f16(
            false, a, false, bf, (short)0, acc[b][t], false, false);
      }
    __syncthreads();
  }
  // C/D layout: VGPR r -> row (r + 8*group), col = lane&15
#pragma unroll
  for (int b = 0; b < BB; ++b)
#pragma unroll
    for (int t = 0; t < NT; ++t)
#pragma unroll
      for (int r = 0; r < 8; ++r) {
        int m = m0 + r + 8 * group;
        if (m < M) Yp[(long)b * M * N + (long)m * N + t * 16 + mrow] = acc[b][t][r];
      }
}

// Deterministic fixed-order reduction of split-K partials (+ optional bias
// broadcast over the last dimension N).
__global__ void __launch_bounds__(256)
reduce_k(const float* __restrict__ part, const float* __restrict__ bias,
         float* __restrict__ out, long chunk, int S, int N) {
  long i = (long)blockIdx.x * 256 + threadIdx.x;
  if (i >= chunk) return;
  float a = bias ? bias[(int)(i % N)] : 0.f;
  for (int s = 0; s < S; ++s) a += part[(long)s * chunk + i];
  out[i] = a;
}

// ---------------------------------------------------------------------------
// PAI conv: one wave handles 16 points of one batch.
//   H[p][t*f+fi] = act( sum_k x[nbr[p][k]][fi] * adjw[p][k][t] )   (LDS, f16)
//   out = act(H @ convW + convb); out[last]=0; res = x @ mlpW + mlpb
// Both GEMMs run on WMMA with K padded to 32.
// ---------------------------------------------------------------------------
extern __shared__ __align__(16) _Float16 conv_smem[];

template <int OUT_PAD, int USE_ELU>
__global__ void __launch_bounds__(32)
pai_conv(const float* __restrict__ x, const int* __restrict__ nbr,
         const float* __restrict__ adjw,
         const float* __restrict__ convW, const float* __restrict__ convb,
         const float* __restrict__ mlpW, const float* __restrict__ mlpb,
         float* __restrict__ y,
         int npts, int fin, int out_c, int kf_pad, int fr_pad) {
  constexpr int NT = OUT_PAD / 16;
  _Float16* Hs = conv_smem;                 // 16 x kf_pad
  _Float16* Xr = Hs + 16 * kf_pad;          // 16 x fr_pad  (residual input)
  _Float16* Ws = Xr + 16 * fr_pad;          // OUT_PAD x 32 (transposed W tile)

  const int lane  = threadIdx.x;
  const int group = lane >> 4;
  const int mrow  = lane & 15;
  const int b     = blockIdx.y;
  const int p0    = blockIdx.x * 16;
  const int KF    = 10 * fin;
  const float* xb = x + (long)b * npts * fin;

  // zero-init (covers K padding of both WMMA operands)
  for (int i = lane; i < 16 * kf_pad; i += 32) Hs[i] = (_Float16)0.f;
  for (int i = lane; i < 16 * fr_pad; i += 32) Xr[i] = (_Float16)0.f;
  __syncthreads();

  // gather + adjacency mix + activation -> H; also stage residual rows
  for (int pair = lane; pair < 16 * fin; pair += 32) {
    int p  = pair / fin;
    int fi = pair - p * fin;
    int gp = p0 + p;
    if (gp < npts) {
      float xv[10];
#pragma unroll
      for (int k = 0; k < 10; ++k) {
        int idx = nbr[gp * 10 + k];
        xv[k] = (idx == npts - 1) ? 0.f : xb[(long)idx * fin + fi];  // padded row is zero
      }
      const float* aw = adjw + (long)gp * 100;
#pragma unroll
      for (int t = 0; t < 10; ++t) {
        float s = 0.f;
#pragma unroll
        for (int k = 0; k < 10; ++k) s += xv[k] * aw[k * 10 + t];
        if (USE_ELU) s = elu_f(s);
        Hs[p * kf_pad + t * fin + fi] = (_Float16)s;
      }
      float xr = (gp == npts - 1) ? 0.f : xb[(long)gp * fin + fi];
      Xr[p * fr_pad + fi] = (_Float16)xr;
    }
  }
  __syncthreads();

  v8f zero = {};
  v8f acc[NT], racc[NT];
#pragma unroll
  for (int t = 0; t < NT; ++t) { acc[t] = zero; racc[t] = zero; }

  // conv GEMM: H(16 x KF) @ convW(KF x out_c)
  for (int kk = 0; kk < kf_pad; kk += 32) {
    for (int i = lane; i < 32 * OUT_PAD; i += 32) {
      int r = i / OUT_PAD, o = i - r * OUT_PAD;
      float v = (o < out_c && (kk + r) < KF) ? convW[(long)(kk + r) * out_c + o] : 0.f;
      Ws[o * 32 + r] = (_Float16)v;
    }
    __syncthreads();
    v16h a = ld_frag(&Hs[mrow * kf_pad + kk], group);
#pragma unroll
    for (int t = 0; t < NT; ++t) {
      v16h bf = ld_frag(&Ws[(t * 16 + mrow) * 32], group);
      acc[t] = __builtin_amdgcn_wmma_f32_16x16x32_f16(
          false, a, false, bf, (short)0, acc[t], false, false);
    }
    __syncthreads();
  }
  // residual GEMM: Xr(16 x fin) @ mlpW(fin x out_c)
  for (int kk = 0; kk < fr_pad; kk += 32) {
    for (int i = lane; i < 32 * OUT_PAD; i += 32) {
      int r = i / OUT_PAD, o = i - r * OUT_PAD;
      float v = (o < out_c && (kk + r) < fin) ? mlpW[(long)(kk + r) * out_c + o] : 0.f;
      Ws[o * 32 + r] = (_Float16)v;
    }
    __syncthreads();
    v16h a = ld_frag(&Xr[mrow * fr_pad + kk], group);
#pragma unroll
    for (int t = 0; t < NT; ++t) {
      v16h bf = ld_frag(&Ws[(t * 16 + mrow) * 32], group);
      racc[t] = __builtin_amdgcn_wmma_f32_16x16x32_f16(
          false, a, false, bf, (short)0, racc[t], false, false);
    }
    __syncthreads();
  }
  // epilogue: bias + act + zero padded row + residual
#pragma unroll
  for (int t = 0; t < NT; ++t) {
    int o = t * 16 + mrow;
    if (o < out_c) {
      float cb = convb[o], mb = mlpb[o];
#pragma unroll
      for (int r = 0; r < 8; ++r) {
        int gp = p0 + r + 8 * group;
        if (gp < npts) {
          float cv = acc[t][r] + cb;
          if (USE_ELU) cv = elu_f(cv);
          if (gp == npts - 1) cv = 0.f;        // out[:, -1, :] = 0
          y[((long)b * npts + gp) * out_c + o] = cv + racc[t][r] + mb;
        }
      }
    }
  }
}

// ---------------------------------------------------------------------------
// Small helper kernels
// ---------------------------------------------------------------------------
__global__ void build_x(const float* __restrict__ x1, const float* __restrict__ x2,
                        const float* __restrict__ x3, float* __restrict__ out) {
  int i = blockIdx.x * 256 + threadIdx.x;
  const int total = 3 * 16001 * 3;
  if (i >= total) return;
  int b = i / (16001 * 3);
  int rem = i - b * (16001 * 3);
  int n = rem / 3, c = rem - n * 3;
  const float* s = (b == 0) ? x1 : ((b == 1) ? x2 : x3);
  out[i] = (n < 16000) ? s[n * 3 + c] : 0.f;
}

__global__ void zcomb(const float* __restrict__ z, float* __restrict__ z0) {
  int o = threadIdx.x;
  z0[o] = z[512 + o] + z[256 + o] - z[o];   // z[2] + (z[1] - z[0])
}

__global__ void __launch_bounds__(256)
fc_dec_k(const float* __restrict__ z0, const float* __restrict__ W,
         const float* __restrict__ bias, float* __restrict__ y) {
  __shared__ float zs[256];
  int t = threadIdx.x;
  zs[t] = z0[t];
  __syncthreads();
  int m = blockIdx.x * 256 + t;
  if (m >= 32128) return;
  float acc = bias[m];
  for (int j = 0; j < 256; ++j) acc += zs[j] * W[(long)j * 32128 + m];
  y[m] = acc;
}

// ---------------------------------------------------------------------------
// Host-side dispatch helpers
// ---------------------------------------------------------------------------
static inline void pick_split(int M, int Kd, int* ks_out, int* S_out) {
  int mt = (M + 15) / 16;
  int target = 2048;                       // waves needed to saturate HBM/latency
  int S = (target + mt - 1) / mt;
  if (S < 1) S = 1;
  int maxS = (Kd + 31) / 32;
  if (S > maxS) S = maxS;
  int ks = ((Kd + S - 1) / S + 31) / 32 * 32;
  S = (Kd + ks - 1) / ks;
  *ks_out = ks; *S_out = S;
}

template <int BB, int NT>
static inline void run_pool(const float* Lm, const float* X, float* Y,
                            float* part, int M, int Kd, hipStream_t stream) {
  constexpr int N = NT * 16;
  int ks, S;
  pick_split(M, Kd, &ks, &S);
  dim3 grid((M + 15) / 16, S);
  if (S == 1) {
    pool_gemm<BB, NT><<<grid, 32, 0, stream>>>(Lm, X, Y, M, Kd, ks);
  } else {
    pool_gemm<BB, NT><<<grid, 32, 0, stream>>>(Lm, X, part, M, Kd, ks);
    long chunk = (long)BB * M * N;
    reduce_k<<<(int)((chunk + 255) / 256), 256, 0, stream>>>(
        part, nullptr, Y, chunk, S, N);
  }
}

static inline void launch_pool(const float* Lm, const float* X, float* Y,
                               float* part, int M, int Kd, int N, int B,
                               hipStream_t stream) {
  if (B == 3 && N == 64)        run_pool<3, 4>(Lm, X, Y, part, M, Kd, stream);
  else if (B == 3 && N == 128)  run_pool<3, 8>(Lm, X, Y, part, M, Kd, stream);
  else if (B == 1 && N == 128)  run_pool<1, 8>(Lm, X, Y, part, M, Kd, stream);
  else                          run_pool<1, 4>(Lm, X, Y, part, M, Kd, stream);
}

static inline void launch_conv(const float* x, const int* nbr, const float* aw,
                               const float* cW, const float* cb,
                               const float* mW, const float* mb,
                               float* y, int npts, int B, int fin, int out_c,
                               bool elu, hipStream_t stream) {
  int kf_pad = ((10 * fin + 31) / 32) * 32;
  int fr_pad = ((fin + 31) / 32) * 32;
  int out_pad = (out_c <= 16) ? 16 : ((out_c <= 64) ? 64 : 128);
  size_t smem = (size_t)(16 * kf_pad + 16 * fr_pad + out_pad * 32) * 2;  // f16 bytes
  dim3 grid((npts + 15) / 16, B);
  dim3 blk(32);
  if (out_pad == 16) {
    if (elu) pai_conv<16, 1><<<grid, blk, smem, stream>>>(x, nbr, aw, cW, cb, mW, mb, y, npts, fin, out_c, kf_pad, fr_pad);
    else     pai_conv<16, 0><<<grid, blk, smem, stream>>>(x, nbr, aw, cW, cb, mW, mb, y, npts, fin, out_c, kf_pad, fr_pad);
  } else if (out_pad == 64) {
    if (elu) pai_conv<64, 1><<<grid, blk, smem, stream>>>(x, nbr, aw, cW, cb, mW, mb, y, npts, fin, out_c, kf_pad, fr_pad);
    else     pai_conv<64, 0><<<grid, blk, smem, stream>>>(x, nbr, aw, cW, cb, mW, mb, y, npts, fin, out_c, kf_pad, fr_pad);
  } else {
    if (elu) pai_conv<128, 1><<<grid, blk, smem, stream>>>(x, nbr, aw, cW, cb, mW, mb, y, npts, fin, out_c, kf_pad, fr_pad);
    else     pai_conv<128, 0><<<grid, blk, smem, stream>>>(x, nbr, aw, cW, cb, mW, mb, y, npts, fin, out_c, kf_pad, fr_pad);
  }
}

extern "C" void kernel_launch(void* const* d_in, const int* in_sizes, int n_in,
                              void* d_out, int out_size, void* d_ws, size_t ws_size,
                              hipStream_t stream) {
  (void)in_sizes; (void)n_in; (void)out_size; (void)ws_size;

  // Input leaves in setup_inputs() dict/pytree order:
  const float* x1 = (const float*)d_in[0];
  const float* x2 = (const float*)d_in[1];
  const float* x3 = (const float*)d_in[2];
  // enc_params[i] = (convW, convb, mlpW, mlpb, adjw) at 3+5i; dec_params[i] at 18+5i
  const float* fceW = (const float*)d_in[38];
  const float* fceB = (const float*)d_in[39];
  const float* fcdW = (const float*)d_in[40];
  const float* fcdB = (const float*)d_in[41];
  const float* D0 = (const float*)d_in[42];
  const float* D1 = (const float*)d_in[43];
  const float* D2 = (const float*)d_in[44];
  const float* U0 = (const float*)d_in[45];
  const float* U1 = (const float*)d_in[46];
  const float* U2 = (const float*)d_in[47];
  const int* nb0 = (const int*)d_in[48];
  const int* nb1 = (const int*)d_in[49];
  const int* nb2 = (const int*)d_in[50];

  #define P(base, j) ((const float*)d_in[(base) + (j)])

  // Workspace: two ping-pong activation buffers + latent + split-K partials.
  const long BUF = 3L * 16001 * 64;        // 3,072,192 floats
  const long PART = 7000000;               // >= max split-K partial footprint
  float* bufA  = (float*)d_ws;
  float* bufB  = bufA + BUF;
  float* zbuf  = bufB + BUF;               // 3*256
  float* z0buf = zbuf + 768;               // 256
  float* partB = z0buf + 256;              // PART floats

  // 1) pad+stack inputs -> (3,16001,3)
  build_x<<<(3 * 16001 * 3 + 255) / 256, 256, 0, stream>>>(x1, x2, x3, bufA);

  // Encoder
  launch_conv(bufA, nb0, P(3, 4), P(3, 0), P(3, 1), P(3, 2), P(3, 3),
              bufB, 16001, 3, 3, 64, true, stream);
  launch_pool(D0, bufB, bufA, partB, 4001, 16001, 64, 3, stream);
  launch_conv(bufA, nb1, P(8, 4), P(8, 0), P(8, 1), P(8, 2), P(8, 3),
              bufB, 4001, 3, 64, 64, true, stream);
  launch_pool(D1, bufB, bufA, partB, 1001, 4001, 64, 3, stream);
  launch_conv(bufA, nb2, P(13, 4), P(13, 0), P(13, 1), P(13, 2), P(13, 3),
              bufB, 1001, 3, 64, 128, true, stream);
  launch_pool(D2, bufB, bufA, partB, 251, 1001, 128, 3, stream);

  // Latent: fc_enc as WMMA GEMM  z(3x256) = Xf(3x32128) @ W(32128x256) + b
  {
    int M = 3, Kd = 32128, N = 256;
    int ks = 32, S = (Kd + ks - 1) / ks;   // 1004-way split-K
    dim3 grid((M + 15) / 16, S);
    pool_gemm<1, 16><<<grid, 32, 0, stream>>>(bufA, fceW, partB, M, Kd, ks);
    long chunk = (long)M * N;
    reduce_k<<<(int)((chunk + 255) / 256), 256, 0, stream>>>(
        partB, fceB, zbuf, chunk, S, N);
  }
  zcomb<<<1, 256, 0, stream>>>(zbuf, z0buf);
  fc_dec_k<<<126, 256, 0, stream>>>(z0buf, fcdW, fcdB, bufB);  // (1,251,128)

  // Decoder
  launch_pool(U2, bufB, bufA, partB, 1001, 251, 128, 1, stream);
  launch_conv(bufA, nb2, P(18, 4), P(18, 0), P(18, 1), P(18, 2), P(18, 3),
              bufB, 1001, 1, 128, 64, true, stream);
  launch_pool(U1, bufB, bufA, partB, 4001, 1001, 64, 1, stream);
  launch_conv(bufA, nb1, P(23, 4), P(23, 0), P(23, 1), P(23, 2), P(23, 3),
              bufB, 4001, 1, 64, 64, true, stream);
  launch_pool(U0, bufB, bufA, partB, 16001, 4001, 64, 1, stream);
  launch_conv(bufA, nb0, P(28, 4), P(28, 0), P(28, 1), P(28, 2), P(28, 3),
              bufB, 16001, 1, 64, 64, true, stream);
  // Final conv (identity activation) writes (16001,3) directly to d_out.
  launch_conv(bufB, nb0, P(33, 4), P(33, 0), P(33, 1), P(33, 2), P(33, 3),
              (float*)d_out, 16001, 1, 64, 3, false, stream);

  #undef P
}
